// QuantumTransformerDual_65481071405669
// MI455X (gfx1250) — compile-verified, matrix operand order checked by reading, not run on hardware
//
#include <hip/hip_runtime.h>
#include <hip/hip_bf16.h>
#include <math.h>

typedef __attribute__((ext_vector_type(16))) _Float16 v16h;
typedef __attribute__((ext_vector_type(8)))  float    v8f;

constexpr int kB  = 16;
constexpr int kT  = 2048;
constexpr int kBT = kB * kT;       // 32768 positions
constexpr int kD  = 512;
constexpr int kH  = 16;
constexpr int kHD = 32;
constexpr int kF  = 2048;
constexpr int kL  = 4;
constexpr int kC  = 10;

// ---------------------------------------------------------------------------
// WMMA fragment helpers (16-bit A/B layout, wave32):
//   A 16x32: lane L holds row M=L%16; lanes 0-15 carry K {0..7,16..23},
//   lanes 16-31 carry K {8..15,24..31}; VGPR j<4 -> K=2j(+kb), j>=4 -> 16+2(j-4)(+kb).
//   B assumed mirrored with N=L%16 (both operands are K-contiguous rows here).
//   Per lane this is two contiguous 16-byte runs -> compiler emits 2x ds_load_b128.
// ---------------------------------------------------------------------------
__device__ __forceinline__ int wmma_k(int lane, int j, int hh) {
  int kb = (lane < 16) ? 0 : 8;
  return (j < 4) ? (2 * j + hh + kb) : (16 + 2 * (j - 4) + hh + kb);
}

__device__ __forceinline__ v16h load_frag_row(const _Float16* row, int lane) {
  v16h f;
#pragma unroll
  for (int e = 0; e < 16; ++e) f[e] = row[wmma_k(lane, e >> 1, e & 1)];
  return f;
}

// ---------------------------------------------------------------------------
// Tiny per-layer scalar prep: rxsum[l] = sum(rx[l,:]); cosry[l] = cos(sum(ry[l,:]))
// ---------------------------------------------------------------------------
__global__ void scalars_kernel(const float* __restrict__ rx, const float* __restrict__ ry,
                               float* __restrict__ rxsum, float* __restrict__ cosry) {
  if (threadIdx.x == 0 && blockIdx.x == 0) {
    for (int l = 0; l < kL; ++l) {
      float s = 0.f;
      for (int i = 0; i < kHD; ++i) s += rx[l * kHD + i];
      rxsum[l] = s;
      float s2 = 0.f;
      for (int i = 0; i < kD; ++i) s2 += ry[l * kD + i];
      cosry[l] = __cosf(s2);
    }
  }
}

__global__ void conv_kernel(const float* __restrict__ src, _Float16* __restrict__ dst, int n) {
  int i = blockIdx.x * blockDim.x + threadIdx.x;
  if (i < n) dst[i] = (_Float16)src[i];
}

__global__ void zero_kernel(float* __restrict__ p, int n) {
  int i = blockIdx.x * blockDim.x + threadIdx.x;
  if (i < n) p[i] = 0.f;
}

// ---------------------------------------------------------------------------
// x[p, :] = emb[tokens[p], :] + pos_encoding(t = p % T)
// ---------------------------------------------------------------------------
__global__ void __launch_bounds__(256) embed_kernel(const int* __restrict__ tokens,
                                                    const float* __restrict__ emb,
                                                    float* __restrict__ x) {
  const int p = blockIdx.x;
  const int t = p % kT;
  const int i = threadIdx.x;  // pair index 0..255
  const int tok = tokens[p];
  const float div = __expf((float)(2 * i) * (-9.210340371976184f / (float)kD));
  float sv, cv;
  __sincosf((float)t * div, &sv, &cv);
  x[(size_t)p * kD + 2 * i]     = emb[(size_t)tok * kD + 2 * i]     + sv;
  x[(size_t)p * kD + 2 * i + 1] = emb[(size_t)tok * kD + 2 * i + 1] + cv;
}

// ---------------------------------------------------------------------------
// LayerNorm + cosine encoder, f16 output.
// mode 0: out = cos(h + rxsum)     mode 1: out = cosry * cos(h)
// ---------------------------------------------------------------------------
__global__ void __launch_bounds__(256) ln_cos_kernel(const float* __restrict__ x,
                                                     const float* __restrict__ s,
                                                     const float* __restrict__ bshift,
                                                     const float* __restrict__ scal,
                                                     int mode,
                                                     _Float16* __restrict__ out) {
  __shared__ float r1[256];
  __shared__ float r2[256];
  const int tid = threadIdx.x;
  const size_t row = blockIdx.x;
  float v0 = x[row * kD + tid];
  float v1 = x[row * kD + tid + 256];
  r1[tid] = v0 + v1;
  r2[tid] = v0 * v0 + v1 * v1;
  __syncthreads();
  for (int st = 128; st > 0; st >>= 1) {
    if (tid < st) { r1[tid] += r1[tid + st]; r2[tid] += r2[tid + st]; }
    __syncthreads();
  }
  const float mu  = r1[0] * (1.0f / kD);
  const float var = r2[0] * (1.0f / kD) - mu * mu;
  const float rs  = rsqrtf(var + 1e-5f);
  const float sc  = *scal;
  float h0 = (v0 - mu) * rs * s[tid]       + bshift[tid];
  float h1 = (v1 - mu) * rs * s[tid + 256] + bshift[tid + 256];
  if (mode == 0) {
    out[row * kD + tid]       = (_Float16)__cosf(h0 + sc);
    out[row * kD + tid + 256] = (_Float16)__cosf(h1 + sc);
  } else {
    out[row * kD + tid]       = (_Float16)(sc * __cosf(h0));
    out[row * kD + tid + 256] = (_Float16)(sc * __cosf(h1));
  }
}

// ---------------------------------------------------------------------------
// Per-position head-mixing attention. One wave per position.
//   scores = qh @ qh^T   -> one V_WMMA_F32_16X16X32_F16 (Gram, A==B fragment)
//   softmax: scores -> LDS; 16 lanes each own one full row (contiguous b128
//            reads) instead of ds_bpermute reductions.
//   out = attn @ qh      -> two K-padded WMMAs (d-halves)
// ---------------------------------------------------------------------------
__global__ void __launch_bounds__(256) attn_kernel(const _Float16* __restrict__ qkv,
                                                   _Float16* __restrict__ attn_out) {
  __shared__ _Float16 sQ[8][kD];   // per-wave qkv row (16 heads x 32 dims)
  __shared__ float    sS[8][256];  // per-wave raw scores 16x16
  __shared__ _Float16 sA[8][256];  // per-wave attn 16x16 (f16, A-operand source)
  const int tid  = threadIdx.x;
  const int w    = tid >> 5;
  const int lane = tid & 31;
  const int rsel = lane & 15;
  const int pos  = blockIdx.x * 8 + w;

  // stage the 512-half row: 16 halves (32B) per lane
  {
    const uint4* src = (const uint4*)(qkv + (size_t)pos * kD + lane * 16);
    uint4* dst = (uint4*)&sQ[w][lane * 16];
    dst[0] = src[0];
    dst[1] = src[1];
  }
  __syncthreads();

  // Gram: A and B fragments are identical (row m == col n == lane%16)
  v16h qa = load_frag_row(&sQ[w][rsel * kHD], lane);
  v8f sc;
#pragma unroll
  for (int e = 0; e < 8; ++e) sc[e] = 0.f;
  sc = __builtin_amdgcn_wmma_f32_16x16x32_f16(false, qa, false, qa, (short)0, sc,
                                              false, false);

  const float scale = 0.17677669529663687f;  // 1/sqrt(32)
#pragma unroll
  for (int r = 0; r < 8; ++r) {
    int h = (lane < 16) ? r : (r + 8);  // C/D layout: VGPR r holds rows r / r+8
    sS[w][h * 16 + rsel] = sc[r] * scale;
  }
  __syncthreads();

  // row-serial softmax: lane g (<16) owns row g of the 16x16 score matrix
  if (lane < 16) {
    float rowv[16];
#pragma unroll
    for (int g = 0; g < 16; ++g) rowv[g] = sS[w][lane * 16 + g];
    float m = rowv[0];
#pragma unroll
    for (int g = 1; g < 16; ++g) m = fmaxf(m, rowv[g]);
    float ssum = 0.f;
#pragma unroll
    for (int g = 0; g < 16; ++g) {
      rowv[g] = __expf(rowv[g] - m);
      ssum += rowv[g];
    }
    float inv = 1.0f / ssum;
#pragma unroll
    for (int g = 0; g < 16; ++g) sA[w][lane * 16 + g] = (_Float16)(rowv[g] * inv);
  }
  __syncthreads();

  // out = attn(16x16, K zero-padded to 32) @ qh(16x32), two N=16 halves
  v16h af;
#pragma unroll
  for (int e = 0; e < 16; ++e) {
    int k = wmma_k(lane, e >> 1, e & 1);
    af[e] = (k < 16) ? sA[w][rsel * 16 + k] : (_Float16)0.0f;
  }
#pragma unroll
  for (int hf = 0; hf < 2; ++hf) {
    v16h bf;
#pragma unroll
    for (int e = 0; e < 16; ++e) {
      int k = wmma_k(lane, e >> 1, e & 1);
      bf[e] = (k < 16) ? sQ[w][k * kHD + hf * 16 + rsel] : (_Float16)0.0f;
    }
    v8f o;
#pragma unroll
    for (int e = 0; e < 8; ++e) o[e] = 0.f;
    o = __builtin_amdgcn_wmma_f32_16x16x32_f16(false, af, false, bf, (short)0, o,
                                               false, false);
#pragma unroll
    for (int r = 0; r < 8; ++r) {
      int h = (lane < 16) ? r : (r + 8);
      attn_out[(size_t)pos * kD + h * kHD + hf * 16 + rsel] = (_Float16)o[r];
    }
  }
}

// ---------------------------------------------------------------------------
// C[M,N] = epilogue( A[M,K] @ W[N,K]^T )
// MODE 0: f32 out + residual     MODE 1: f16 out + ReLU
// Block tile 128x128, 8 waves (4m x 2n), wave tile 32x64 = 2x4 WMMA tiles.
// Double-buffered LDS (2x 16KB); global loads for tile k+1 staged in registers
// while tile k computes; one barrier per K-step; prefetch of tile k+2.
// ---------------------------------------------------------------------------
template <int MODE>
__global__ void __launch_bounds__(256) gemm_kernel(const _Float16* __restrict__ A,
                                                   const _Float16* __restrict__ W,
                                                   const float* __restrict__ resIn,
                                                   float* __restrict__ outF,
                                                   _Float16* __restrict__ outH,
                                                   int M, int N, int K) {
  __shared__ _Float16 As[2][128 * 32];
  __shared__ _Float16 Ws[2][128 * 32];
  const int tid   = threadIdx.x;
  const int lane  = tid & 31;
  const int wave  = tid >> 5;
  const int rsel  = lane & 15;
  const int mTile = blockIdx.y * 128;
  const int nTile = blockIdx.x * 128;
  const int mBase = (wave >> 1) * 32;   // 4 waves over M
  const int nBase = (wave & 1) * 64;    // 2 waves over N

  v8f acc[2][4];
#pragma unroll
  for (int i = 0; i < 2; ++i)
#pragma unroll
    for (int j = 0; j < 4; ++j)
#pragma unroll
      for (int e = 0; e < 8; ++e) acc[i][j][e] = 0.f;

  const int row = tid >> 1;   // 128 rows
  const int seg = tid & 1;    // 2 x 16 halves per row
  const int ktiles = K >> 5;

  uint4 ra0, ra1, rw0, rw1;
  auto loadRegs = [&](int kt) {
    const uint4* ap = (const uint4*)(A + (size_t)(mTile + row) * K + kt * 32 + seg * 16);
    ra0 = ap[0];
    ra1 = ap[1];
    const uint4* wp = (const uint4*)(W + (size_t)(nTile + row) * K + kt * 32 + seg * 16);
    rw0 = wp[0];
    rw1 = wp[1];
  };
  auto storeLDS = [&](int buf) {
    uint4* ad = (uint4*)&As[buf][row * 32 + seg * 16];
    ad[0] = ra0;
    ad[1] = ra1;
    uint4* wd = (uint4*)&Ws[buf][row * 32 + seg * 16];
    wd[0] = rw0;
    wd[1] = rw1;
  };

  loadRegs(0);
  storeLDS(0);
  for (int kt = 0; kt < ktiles; ++kt) {
    const int cur = kt & 1;
    if (kt + 1 < ktiles) loadRegs(kt + 1);
    if (kt + 2 < ktiles) {  // global_prefetch_b8 of tile k+2
      __builtin_prefetch(A + (size_t)(mTile + row) * K + (kt + 2) * 32 + seg * 16, 0, 1);
      __builtin_prefetch(W + (size_t)(nTile + row) * K + (kt + 2) * 32 + seg * 16, 0, 1);
    }
    __syncthreads();  // LDS[cur] produced by previous iteration is visible

    v16h af[2], bf[4];
#pragma unroll
    for (int i = 0; i < 2; ++i)
      af[i] = load_frag_row(&As[cur][(mBase + i * 16 + rsel) * 32], lane);
#pragma unroll
    for (int j = 0; j < 4; ++j)
      bf[j] = load_frag_row(&Ws[cur][(nBase + j * 16 + rsel) * 32], lane);
#pragma unroll
    for (int i = 0; i < 2; ++i)
#pragma unroll
      for (int j = 0; j < 4; ++j)
        acc[i][j] = __builtin_amdgcn_wmma_f32_16x16x32_f16(
            false, af[i], false, bf[j], (short)0, acc[i][j], false, false);

    if (kt + 1 < ktiles) storeLDS(cur ^ 1);  // safe: nothing reads buf cur^1 now
  }

#pragma unroll
  for (int i = 0; i < 2; ++i)
#pragma unroll
    for (int j = 0; j < 4; ++j)
#pragma unroll
      for (int r = 0; r < 8; ++r) {
        int m = mTile + mBase + i * 16 + ((lane < 16) ? r : (r + 8));
        int n = nTile + nBase + j * 16 + rsel;
        float v = acc[i][j][r];
        if (MODE == 0) {
          outF[(size_t)m * N + n] = resIn[(size_t)m * N + n] + v;
        } else {
          outH[(size_t)m * N + n] = (_Float16)fmaxf(v, 0.0f);
        }
      }
}

// ---------------------------------------------------------------------------
// pooled[b,:] += partial sums over a 32-row T slice (global_atomic_add_f32)
// ---------------------------------------------------------------------------
__global__ void __launch_bounds__(256) pool_partial_kernel(const float* __restrict__ x,
                                                           float* __restrict__ pooled) {
  const int b  = blockIdx.x;   // 16
  const int ts = blockIdx.y;   // 64 slices of 32 rows
  const int tid = threadIdx.x;
  float a0 = 0.f, a1 = 0.f;
  for (int tt = 0; tt < 32; ++tt) {
    const float* rowp = x + ((size_t)b * kT + ts * 32 + tt) * kD;
    a0 += rowp[tid];
    a1 += rowp[tid + 256];
  }
  atomicAdd(&pooled[b * kD + tid], a0);
  atomicAdd(&pooled[b * kD + tid + 256], a1);
}

__global__ void clf_kernel(const float* __restrict__ pooled,
                           const float* __restrict__ clf_w,
                           const float* __restrict__ clf_b,
                           float* __restrict__ out) {
  const int t = threadIdx.x;
  if (t < kB * kC) {
    const int b = t / kC, c = t % kC;
    float s = 0.f;
    for (int d = 0; d < kD; ++d) s += pooled[b * kD + d] * clf_w[c * kD + d];
    out[b * kC + c] = clf_b[c] + s * (1.0f / kT);
  }
}

// ---------------------------------------------------------------------------
extern "C" void kernel_launch(void* const* d_in, const int* in_sizes, int n_in,
                              void* d_out, int out_size, void* d_ws, size_t ws_size,
                              hipStream_t stream) {
  const int*   tokens = (const int*)  d_in[0];
  const float* emb    = (const float*)d_in[1];
  const float* ln1_s  = (const float*)d_in[2];
  const float* ln1_b  = (const float*)d_in[3];
  const float* ln2_s  = (const float*)d_in[4];
  const float* ln2_b  = (const float*)d_in[5];
  const float* rx     = (const float*)d_in[6];
  const float* ry     = (const float*)d_in[7];
  const float* proj_w = (const float*)d_in[8];
  const float* lin1_w = (const float*)d_in[9];
  const float* lin2_w = (const float*)d_in[10];
  const float* clf_w  = (const float*)d_in[11];
  const float* clf_b  = (const float*)d_in[12];
  float* out = (float*)d_out;

  char* ws = (char*)d_ws;
  size_t off = 0;
  auto alloc = [&](size_t bytes) -> char* {
    char* p = ws + off;
    off += (bytes + 255) & ~(size_t)255;
    return p;
  };
  float*    x      = (float*)   alloc((size_t)kBT * kD * 4);      //  64 MB activations
  _Float16* qkv    = (_Float16*)alloc((size_t)kBT * kD * 2);      //  32 MB (reused as q_ff)
  _Float16* aout   = (_Float16*)alloc((size_t)kBT * kD * 2);      //  32 MB attention out
  _Float16* ff1    = (_Float16*)alloc((size_t)kBT * kF * 2);      // 128 MB hidden
  _Float16* wP     = (_Float16*)alloc((size_t)kL * kD * kD * 2);  //   2 MB f16 proj_w
  _Float16* w1     = (_Float16*)alloc((size_t)kL * kF * kD * 2);  //   8 MB f16 lin1_w
  _Float16* w2     = (_Float16*)alloc((size_t)kL * kD * kF * 2);  //   8 MB f16 lin2_w
  float*    pooled = (float*)   alloc((size_t)kB * kD * 4);       //  32 KB pooled sums
  float*    rxs    = (float*)   alloc(256);
  float*    cry    = rxs + 8;

  scalars_kernel<<<1, 32, 0, stream>>>(rx, ry, rxs, cry);
  {
    int n = kL * kD * kD;
    conv_kernel<<<(n + 255) / 256, 256, 0, stream>>>(proj_w, wP, n);
    n = kL * kF * kD;
    conv_kernel<<<(n + 255) / 256, 256, 0, stream>>>(lin1_w, w1, n);
    n = kL * kD * kF;
    conv_kernel<<<(n + 255) / 256, 256, 0, stream>>>(lin2_w, w2, n);
  }
  embed_kernel<<<kBT, 256, 0, stream>>>(tokens, emb, x);

  for (int l = 0; l < kL; ++l) {
    // qkv = cos(LN1(x) + sum(rx))
    ln_cos_kernel<<<kBT, 256, 0, stream>>>(x, ln1_s + l * kD, ln1_b + l * kD,
                                           rxs + l, 0, qkv);
    // per-position head-mixing attention (WMMA 16x16x32)
    attn_kernel<<<kBT / 8, 256, 0, stream>>>(qkv, aout);
    // x = x + aout @ proj_w^T
    gemm_kernel<0><<<dim3(kD / 128, kBT / 128), 256, 0, stream>>>(
        aout, wP + (size_t)l * kD * kD, x, x, nullptr, kBT, kD, kD);
    // q_ff = cos(sum(ry)) * cos(LN2(x))   (reuse qkv buffer)
    ln_cos_kernel<<<kBT, 256, 0, stream>>>(x, ln2_s + l * kD, ln2_b + l * kD,
                                           cry + l, 1, qkv);
    // ff1 = relu(q_ff @ lin1_w^T)
    gemm_kernel<1><<<dim3(kF / 128, kBT / 128), 256, 0, stream>>>(
        qkv, w1 + (size_t)l * kF * kD, nullptr, nullptr, ff1, kBT, kF, kD);
    // x = x + ff1 @ lin2_w^T
    gemm_kernel<0><<<dim3(kD / 128, kBT / 128), 256, 0, stream>>>(
        ff1, w2 + (size_t)l * kD * kF, x, x, nullptr, kBT, kD, kF);
  }

  zero_kernel<<<(kB * kD + 255) / 256, 256, 0, stream>>>(pooled, kB * kD);
  pool_partial_kernel<<<dim3(kB, kT / 32), 256, 0, stream>>>(x, pooled);
  clf_kernel<<<1, 192, 0, stream>>>(pooled, clf_w, clf_b, out);
}
